// EGNN_47863115547336
// MI455X (gfx1250) — matrix operand order repaired and checked by reading, not compiled, not run on hardware
//
#include <hip/hip_runtime.h>
#include <math.h>

typedef __attribute__((ext_vector_type(16))) _Float16 v16h;
typedef __attribute__((ext_vector_type(8)))  float    v8f;

#define HID   64
#define NGRP  64
#define DCUT  5.0f
#define GNEPS 1e-5f

// ---------------- WMMA fragment helpers (wave32, 16x16x32 f16) ----------------

__device__ __forceinline__ v8f vzero8() {
  v8f z;
#pragma unroll
  for (int i = 0; i < 8; ++i) z[i] = 0.0f;
  return z;
}

__device__ __forceinline__ v8f wmma_f16(v16h a, v16h b, v8f c) {
  // D = A(16x32 f16) x B(32x16 f16) + C(16x16 f32)
  return __builtin_amdgcn_wmma_f32_16x16x32_f16(false, a, false, b, (short)0, c,
                                                false, false);
}

// A fragment (16x32, f16) from row-major f16 tile with leading dim ld (elements).
// Lane L(<16) = row L, K 0..7 / 16..23 ; lane L+16 = row L, K 8..15 / 24..31.
__device__ __forceinline__ v16h a_frag_lds(const _Float16* tile, int ld) {
  int lane = threadIdx.x & 31;
  const _Float16* p = tile + (lane & 15) * ld + ((lane >> 4) << 3);
  v16h a;
#pragma unroll
  for (int i = 0; i < 8; ++i) a[i] = p[i];
#pragma unroll
  for (int i = 0; i < 8; ++i) a[8 + i] = p[16 + i];
  return a;
}

// B fragment (32x16) computing  B[k][n] = W[o0+n][k0+k]  from row-major W (f16, ld=ldw).
// Per-lane: K 8h..8h+7 / 16+8h..16+8h+7 -> two 16B vector LDS loads.
__device__ __forceinline__ v16h b_frag_lds(const _Float16* w, int ldw, int o0, int k0) {
  typedef __attribute__((ext_vector_type(8))) _Float16 v8h_t;
  int lane = threadIdx.x & 31;
  const _Float16* p = w + (o0 + (lane & 15)) * ldw + k0 + ((lane >> 4) << 3);
  v8h_t lo = *(const v8h_t*)p;
  v8h_t hi = *(const v8h_t*)(p + 16);
  return __builtin_shufflevector(lo, hi, 0, 1, 2, 3, 4, 5, 6, 7, 8, 9, 10, 11, 12, 13, 14, 15);
}

// ---------------- math helpers (branch-free, VALU-lean) ----------------

// tanh(x) = 1 - 2/(exp(2x)+1); exact saturation at +/-inf, no exec-mask branches.
__device__ __forceinline__ float fast_tanh(float x) {
  float t = __expf(2.0f * x);
  return 1.0f - 2.0f * __builtin_amdgcn_rcpf(t + 1.0f);
}

__device__ __forceinline__ float elu(float x) {
  return x > 0.0f ? x : (__expf(x) - 1.0f);
}

// ---------------- weight staging ----------------

__device__ __forceinline__ void stage_h(const float* g, _Float16* l, int n, int tid, int nt) {
  for (int i = tid; i < n; i += nt) l[i] = (_Float16)g[i];
}
__device__ __forceinline__ void stage_f(const float* g, float* l, int n, int tid, int nt) {
  for (int i = tid; i < n; i += nt) l[i] = g[i];
}

// ---------------- embed gather ----------------

__global__ void egnn_embed_kernel(const float* __restrict__ table,
                                  const int* __restrict__ z,
                                  float* __restrict__ x, int N) {
  int i = blockIdx.x * blockDim.x + threadIdx.x;
  if (i >= N * HID) return;
  int n = i >> 6, f = i & 63;
  x[i] = table[z[n] * HID + f];
}

// ---------------- edge / message kernel ----------------
// Per wave: 16 edges. RBF in registers -> WMMA chain -> f32 atomics scatter.

#define EW 8  // waves per block

__global__ __launch_bounds__(256) void egnn_edge_kernel(
    const float* __restrict__ x,      // [N][64]
    const float* __restrict__ dist,   // [E]
    const int*   __restrict__ i_s, const int* __restrict__ i_t,
    const float* __restrict__ w_dist, // [64][64]
    const float* __restrict__ w_edge, // [64][128]
    const float* __restrict__ W1, const float* __restrict__ b1,
    const float* __restrict__ W2, const float* __restrict__ b2,
    float* __restrict__ aggr,         // [N][64]
    int E) {
  __shared__ __align__(32) _Float16 s_wd[64 * 64];
  __shared__ __align__(32) _Float16 s_we[64 * 128];
  __shared__ __align__(32) _Float16 s_w1[64 * 64];
  __shared__ __align__(32) _Float16 s_w2[64 * 64];
  __shared__ float s_b1[64], s_b2[64];
  __shared__ __align__(32) _Float16 s_m[EW][16 * 64];  // wave-private tiles

  int tid = threadIdx.x, nt = blockDim.x;
  stage_h(w_dist, s_wd, 64 * 64, tid, nt);
  stage_h(w_edge, s_we, 64 * 128, tid, nt);
  stage_h(W1, s_w1, 64 * 64, tid, nt);
  stage_h(W2, s_w2, 64 * 64, tid, nt);
  stage_f(b1, s_b1, 64, tid, nt);
  stage_f(b2, s_b2, 64, tid, nt);
  __syncthreads();

  int wave = tid >> 5, lane = tid & 31;
  int e0 = (blockIdx.x * EW + wave) * 16;
  if (e0 >= E) return;  // wave-uniform: EXEC stays all-ones for WMMA

  _Float16* mbuf = s_m[wave];
  int row = lane & 15;
  int kbase = (lane >> 4) << 3;   // A-frag K base (0 or 8)
  int col = lane & 15;            // D-layout column
  int rbase = (lane >> 4) << 3;   // D-layout row base (0 or 8)

  int e = e0 + row;
  int ec = e < E ? e : (E - 1);   // clamped load index (tail safety)
  int src = i_s[ec];
  int tgt = i_t[ec];
  float d = dist[ec];

  // --- RBF features straight into A fragments (no LDS) ---
  const float START = 0.006737946999085467f;  // exp(-5)
  const float SPAN = (1.0f - START) / 63.0f;
  const float C0 = 0.03125f * (1.0f - START);
  const float BETA = 1.0f / (C0 * C0);
  float dc = fminf(d, DCUT);
  float cutoff = 0.5f * (__cosf(dc * (3.14159265358979323846f / DCUT)) + 1.0f);
  float expd = __expf(-d);

  v16h arbf[2];
#pragma unroll
  for (int c = 0; c < 2; ++c) {
#pragma unroll
    for (int i = 0; i < 8; ++i) {
      int k0 = c * 32 + kbase + i;
      int k1 = k0 + 16;
      float t0 = expd - (START + k0 * SPAN);
      float t1 = expd - (START + k1 * SPAN);
      arbf[c][i]     = (_Float16)(cutoff * __expf(-BETA * t0 * t0));
      arbf[c][8 + i] = (_Float16)(cutoff * __expf(-BETA * t1 * t1));
    }
  }

  // --- dist_emb = tanh(rbf @ w_dist^T) ---
  v8f demb[4];
#pragma unroll
  for (int t = 0; t < 4; ++t) {
    v8f acc = vzero8();
    acc = wmma_f16(arbf[0], b_frag_lds(s_wd, 64, t * 16, 0), acc);
    acc = wmma_f16(arbf[1], b_frag_lds(s_wd, 64, t * 16, 32), acc);
#pragma unroll
    for (int v = 0; v < 8; ++v) acc[v] = fast_tanh(acc[v]);
    demb[t] = acc;
  }

  // --- m = dist_emb * ([x_s | x_t] @ w_edge^T)  (K = 128, 4 chunks) ---
  v8f m[4];
#pragma unroll
  for (int t = 0; t < 4; ++t) m[t] = vzero8();
#pragma unroll
  for (int c = 0; c < 4; ++c) {
    const float* xr = x + (size_t)(c < 2 ? src : tgt) * HID + (c & 1) * 32 + kbase;
    v16h a;
#pragma unroll
    for (int i = 0; i < 8; ++i) a[i] = (_Float16)xr[i];
#pragma unroll
    for (int i = 0; i < 8; ++i) a[8 + i] = (_Float16)xr[16 + i];
#pragma unroll
    for (int t = 0; t < 4; ++t)
      m[t] = wmma_f16(a, b_frag_lds(s_we, 128, t * 16, c * 32), m[t]);
  }
#pragma unroll
  for (int t = 0; t < 4; ++t) {
#pragma unroll
    for (int v = 0; v < 8; ++v) m[t][v] *= demb[t][v];
#pragma unroll
    for (int v = 0; v < 8; ++v)
      mbuf[(rbase + v) * 64 + t * 16 + col] = (_Float16)m[t][v];
  }

  // --- m = elu(m @ W1^T + b1) ---
  {
    v16h a0 = a_frag_lds(mbuf, 64);
    v16h a1 = a_frag_lds(mbuf + 32, 64);
#pragma unroll
    for (int t = 0; t < 4; ++t) {
      v8f acc = vzero8();
      acc = wmma_f16(a0, b_frag_lds(s_w1, 64, t * 16, 0), acc);
      acc = wmma_f16(a1, b_frag_lds(s_w1, 64, t * 16, 32), acc);
      float bias = s_b1[t * 16 + col];
#pragma unroll
      for (int v = 0; v < 8; ++v) acc[v] = elu(acc[v] + bias);
      m[t] = acc;
    }
#pragma unroll
    for (int t = 0; t < 4; ++t)
#pragma unroll
      for (int v = 0; v < 8; ++v)
        mbuf[(rbase + v) * 64 + t * 16 + col] = (_Float16)m[t][v];
  }

  // --- m = elu(m @ W2^T + b2); scatter-add to aggr[i_t] ---
  {
    v16h a0 = a_frag_lds(mbuf, 64);
    v16h a1 = a_frag_lds(mbuf + 32, 64);
    int it[8];
    bool ok[8];
#pragma unroll
    for (int v = 0; v < 8; ++v) {
      int er = e0 + rbase + v;
      ok[v] = er < E;
      it[v] = i_t[ok[v] ? er : (E - 1)];
    }
#pragma unroll
    for (int t = 0; t < 4; ++t) {
      v8f acc = vzero8();
      acc = wmma_f16(a0, b_frag_lds(s_w2, 64, t * 16, 0), acc);
      acc = wmma_f16(a1, b_frag_lds(s_w2, 64, t * 16, 32), acc);
      float bias = s_b2[t * 16 + col];
#pragma unroll
      for (int v = 0; v < 8; ++v) {
        float h = elu(acc[v] + bias);
        if (ok[v])
          unsafeAtomicAdd(&aggr[(size_t)it[v] * HID + t * 16 + col], h);
      }
    }
  }
}

// ---------------- node update kernel ----------------
// h = elu(x @ res^T + [x|aggr] @ comb_W^T + comb_b)

__global__ __launch_bounds__(256) void egnn_node_kernel(
    const float* __restrict__ x, const float* __restrict__ aggr,
    const float* __restrict__ res,   // [64][64]
    const float* __restrict__ combW, // [64][128]
    const float* __restrict__ combB, // [64]
    float* __restrict__ h, int N) {
  __shared__ __align__(32) _Float16 s_res[64 * 64];
  __shared__ __align__(32) _Float16 s_cmb[64 * 128];
  __shared__ float s_cb[64];
  int tid = threadIdx.x, nt = blockDim.x;
  stage_h(res, s_res, 64 * 64, tid, nt);
  stage_h(combW, s_cmb, 64 * 128, tid, nt);
  stage_f(combB, s_cb, 64, tid, nt);
  __syncthreads();

  int wave = tid >> 5, lane = tid & 31;
  int n0 = (blockIdx.x * EW + wave) * 16;
  if (n0 >= N) return;
  int row = lane & 15;
  int kbase = (lane >> 4) << 3;
  int col = lane & 15, rbase = (lane >> 4) << 3;
  int nr = n0 + row;
  if (nr >= N) nr = N - 1;  // clamped tail loads

  v16h ax[2], ag[2];
#pragma unroll
  for (int c = 0; c < 2; ++c) {
    const float* xr = x + (size_t)nr * HID + c * 32 + kbase;
    const float* gr = aggr + (size_t)nr * HID + c * 32 + kbase;
#pragma unroll
    for (int i = 0; i < 8; ++i) { ax[c][i] = (_Float16)xr[i]; ax[c][8 + i] = (_Float16)xr[16 + i]; }
#pragma unroll
    for (int i = 0; i < 8; ++i) { ag[c][i] = (_Float16)gr[i]; ag[c][8 + i] = (_Float16)gr[16 + i]; }
  }

#pragma unroll
  for (int t = 0; t < 4; ++t) {
    v8f acc = vzero8();
    acc = wmma_f16(ax[0], b_frag_lds(s_res, 64, t * 16, 0), acc);
    acc = wmma_f16(ax[1], b_frag_lds(s_res, 64, t * 16, 32), acc);
    acc = wmma_f16(ax[0], b_frag_lds(s_cmb, 128, t * 16, 0), acc);
    acc = wmma_f16(ax[1], b_frag_lds(s_cmb, 128, t * 16, 32), acc);
    acc = wmma_f16(ag[0], b_frag_lds(s_cmb, 128, t * 16, 64), acc);
    acc = wmma_f16(ag[1], b_frag_lds(s_cmb, 128, t * 16, 96), acc);
    float bias = s_cb[t * 16 + col];
#pragma unroll
    for (int v = 0; v < 8; ++v) {
      int nn = n0 + rbase + v;
      if (nn < N)
        h[(size_t)nn * HID + t * 16 + col] = elu(acc[v] + bias);
    }
  }
}

// ---------------- GraphNorm ----------------
// batch is sorted: run-length accumulate per thread, one atomic per graph-run.
// Block = 256 threads covers 64 nodes: thread -> feature f = tid&63,
// node group g = tid>>6 handles nodes [n0+g*16, n0+g*16+16).

#define GN_NODES 64

__global__ __launch_bounds__(256) void gn_sum_kernel(
    const float* __restrict__ h, const int* __restrict__ batch,
    float* __restrict__ sum, float* __restrict__ cnt, int N) {
  int f = threadIdx.x & 63;
  int g = threadIdx.x >> 6;
  int nbeg = blockIdx.x * GN_NODES + g * 16;
  int nend = nbeg + 16;
  if (nend > N) nend = N;
  if (nbeg >= nend) return;

  int cur = batch[nbeg];
  float acc = 0.0f, cacc = 0.0f;
  for (int n = nbeg; n < nend; ++n) {
    int b = batch[n];
    if (b != cur) {
      unsafeAtomicAdd(&sum[cur * HID + f], acc);
      if (f == 0) unsafeAtomicAdd(&cnt[cur], cacc);
      cur = b; acc = 0.0f; cacc = 0.0f;
    }
    acc += h[(size_t)n * HID + f];
    cacc += 1.0f;
  }
  unsafeAtomicAdd(&sum[cur * HID + f], acc);
  if (f == 0) unsafeAtomicAdd(&cnt[cur], cacc);
}

__global__ __launch_bounds__(256) void gn_center_kernel(
    float* __restrict__ h, const int* __restrict__ batch,
    const float* __restrict__ sum, const float* __restrict__ cnt,
    const float* __restrict__ ms, float* __restrict__ var, int N) {
  int f = threadIdx.x & 63;
  int g = threadIdx.x >> 6;
  int nbeg = blockIdx.x * GN_NODES + g * 16;
  int nend = nbeg + 16;
  if (nend > N) nend = N;
  if (nbeg >= nend) return;

  float msf = ms[f];
  int cur = batch[nbeg];
  float mean = sum[cur * HID + f] / fmaxf(cnt[cur], 1.0f);
  float vacc = 0.0f;
  for (int n = nbeg; n < nend; ++n) {
    int b = batch[n];
    if (b != cur) {
      unsafeAtomicAdd(&var[cur * HID + f], vacc);
      cur = b; vacc = 0.0f;
      mean = sum[cur * HID + f] / fmaxf(cnt[cur], 1.0f);
    }
    float out = h[(size_t)n * HID + f] - mean * msf;
    h[(size_t)n * HID + f] = out;
    vacc += out * out;
  }
  unsafeAtomicAdd(&var[cur * HID + f], vacc);
}

__global__ void gn_final_kernel(const float* __restrict__ xin, const float* __restrict__ out,
                                const int* __restrict__ batch, const float* __restrict__ var,
                                const float* __restrict__ cnt, const float* __restrict__ w,
                                const float* __restrict__ bb, float* __restrict__ xout, int N) {
  int i = blockIdx.x * blockDim.x + threadIdx.x;
  if (i >= N * HID) return;
  int n = i >> 6, f = i & 63;
  int b = batch[n];
  float c = fmaxf(cnt[b], 1.0f);
  float inv = rsqrtf(var[b * HID + f] / c + GNEPS);
  xout[i] = xin[i] + w[f] * out[i] * inv + bb[f];
}

// ---------------- launch ----------------

struct LayerP {
  const float *w_dist, *w_edge, *W1, *b1, *W2, *b2, *res, *combW, *combB, *gn_w, *gn_b, *gn_ms;
};

extern "C" void kernel_launch(void* const* d_in, const int* in_sizes, int n_in,
                              void* d_out, int out_size, void* d_ws, size_t ws_size,
                              hipStream_t stream) {
  const float* embed = (const float*)d_in[0];
  LayerP L[2];
  int idx = 1;
  for (int l = 0; l < 2; ++l) {
    L[l].w_dist = (const float*)d_in[idx++];
    L[l].w_edge = (const float*)d_in[idx++];
    L[l].W1 = (const float*)d_in[idx++];
    L[l].b1 = (const float*)d_in[idx++];
    L[l].W2 = (const float*)d_in[idx++];
    L[l].b2 = (const float*)d_in[idx++];
    L[l].res = (const float*)d_in[idx++];
    L[l].combW = (const float*)d_in[idx++];
    L[l].combB = (const float*)d_in[idx++];
    L[l].gn_w = (const float*)d_in[idx++];
    L[l].gn_b = (const float*)d_in[idx++];
    L[l].gn_ms = (const float*)d_in[idx++];
  }
  const float* distance = (const float*)d_in[idx];
  int E = in_sizes[idx]; idx++;
  const int* z = (const int*)d_in[idx];
  int N = in_sizes[idx]; idx++;
  const int* edge_index = (const int*)d_in[idx++];
  const int* batch = (const int*)d_in[idx++];
  const int* i_s = edge_index;
  const int* i_t = edge_index + E;

  // workspace layout (floats)
  float* x0   = (float*)d_ws;
  float* x1   = x0 + (size_t)N * HID;
  float* aggr = x1 + (size_t)N * HID;
  float* hbuf = aggr + (size_t)N * HID;
  float* sum  = hbuf + (size_t)N * HID;
  float* var  = sum + NGRP * HID;
  float* cnt  = var + NGRP * HID;

  int NF = N * HID;
  dim3 eg((NF + 255) / 256), eb(256);
  int gnBlocks = (N + GN_NODES - 1) / GN_NODES;

  egnn_embed_kernel<<<eg, eb, 0, stream>>>(embed, z, x0, N);

  const float* xin = x0;
  for (int l = 0; l < 2; ++l) {
    float* xout = (l == 1) ? (float*)d_out : x1;

    hipMemsetAsync(aggr, 0, (size_t)NF * sizeof(float), stream);
    hipMemsetAsync(sum, 0, (size_t)(2 * NGRP * HID + NGRP) * sizeof(float), stream);

    int eblocks = (E + 16 * EW - 1) / (16 * EW);
    egnn_edge_kernel<<<eblocks, 256, 0, stream>>>(
        xin, distance, i_s, i_t, L[l].w_dist, L[l].w_edge, L[l].W1, L[l].b1,
        L[l].W2, L[l].b2, aggr, E);

    int nblocks = (N + 16 * EW - 1) / (16 * EW);
    egnn_node_kernel<<<nblocks, 256, 0, stream>>>(
        xin, aggr, L[l].res, L[l].combW, L[l].combB, hbuf, N);

    gn_sum_kernel<<<gnBlocks, 256, 0, stream>>>(hbuf, batch, sum, cnt, N);
    gn_center_kernel<<<gnBlocks, 256, 0, stream>>>(hbuf, batch, sum, cnt,
                                                   L[l].gn_ms, var, N);
    gn_final_kernel<<<eg, eb, 0, stream>>>(xin, hbuf, batch, var, cnt, L[l].gn_w,
                                           L[l].gn_b, xout, N);
    xin = xout;
  }
}
